// EMA_43834436223344
// MI455X (gfx1250) — matrix operand order repaired and checked by reading, not compiled, not run on hardware
//
#include <hip/hip_runtime.h>
#include <hip/hip_bf16.h>
#include <stdint.h>

// EMA block for MI455X (gfx1250).
// - All GEMMs on v_wmma_f32_16x16x32_bf16 (bf16 operands, f32 accumulate).
// - Shared B tiles staged to LDS with CDNA5 async copies
//   (global_load_async_to_lds_b128 + s_wait_asynccnt), double-buffered.
// - Per-wave A fragments double-buffered in registers; all 8 B fragments of a
//   chunk preloaded into distinct registers so ds_loads clause ahead of WMMAs.
// - z column-normalization dropped (absorbed exactly by m's per-column l2norm).

typedef __bf16 bf16_t;
typedef __attribute__((ext_vector_type(16))) __bf16 v16bf;
typedef __attribute__((ext_vector_type(8)))  float  v8f;

#define CC 512   // channels
#define NN 4096  // pixels (64x64)
#define KK 64    // bases
#define BB 16    // batch
#define STAGES 3

#define KC 64        // K-chunk staged per step (two 32-K WMMA steps)
#define BSTRIDE 72   // padded LDS row stride (elements): 144B -> conflict-free
#define LDSB_ELEMS (64 * BSTRIDE)

__device__ __forceinline__ bf16_t f2bf(float f) {
  union { float f; uint32_t u; } v; v.f = f;
  uint32_t r = (v.u + 0x7FFFu + ((v.u >> 16) & 1u)) >> 16;  // RNE
  union { uint16_t s; bf16_t b; } o; o.s = (uint16_t)r;
  return o.b;
}

__device__ __forceinline__ v8f vzero8() {
  v8f z = {0.f,0.f,0.f,0.f,0.f,0.f,0.f,0.f};
  return z;
}

__device__ __forceinline__ unsigned lds_off32(const void* p) {
  // generic pointer to LDS: low 32 bits carry the LDS byte offset
  return (unsigned)(uintptr_t)p;
}

__device__ __forceinline__ void async_wait0() {
  asm volatile("s_wait_asynccnt 0x0" ::: "memory");
}

// Async-stage a 64(N) x 64(K) bf16 tile of Bt (N,K row-major) into LDS
// (rows padded to BSTRIDE). 256 threads x 2 ops x 16B = 8KB tile.
__device__ __forceinline__ void async_copy_B(const bf16_t* __restrict__ Bt,
                                             int ldb, int col0, int kc,
                                             bf16_t* ldsbuf) {
  const int tid = threadIdx.x;
  const int u4  = tid & 7;        // 16B segment within a 64-elem row
#pragma unroll
  for (int i = 0; i < 2; ++i) {
    const int row = (tid >> 3) + 32 * i;
    const bf16_t* g = Bt + (size_t)(col0 + row) * ldb + kc + u4 * 8;
    unsigned l = lds_off32(ldsbuf + row * BSTRIDE + u4 * 8);
    asm volatile("global_load_async_to_lds_b128 %0, %1, off"
                 :: "v"(l), "v"(g) : "memory");
  }
}

// ---------------------------------------------------------------------------
// Block-level GEMM core: 8 waves, block tile 128(M) x 64(N); each wave owns a
// 16x64 strip (4 accumulators). A: (M,K) bf16 row-major (per-wave, register
// double-buffered). Bt: (N,K) bf16 row-major (block-shared, LDS
// double-buffered via async copies). Fragment element order matches the ISA
// 16-bit A(16x32)/B(32x16) VGPR layouts (lane half selects the K-subgroup).
// ---------------------------------------------------------------------------
__device__ __forceinline__ void gemm_block(const bf16_t* __restrict__ A, int lda,
                                           const bf16_t* __restrict__ Bt, int ldb,
                                           int Ksize, int row0, int col0,
                                           v8f acc[4], bf16_t* ldsB) {
  const int lane = threadIdx.x & 31;
  const int half = lane >> 4;
  const int l16  = lane & 15;
  const bf16_t* pa0 = A + (size_t)(row0 + l16) * lda + half * 8;

  // Prologue: stage chunk 0 (B -> LDS async, A -> regs)
  async_copy_B(Bt, ldb, col0, 0, ldsB);
  uint4 a_cur[4], a_nxt[4];
#pragma unroll
  for (int j = 0; j < 4; ++j) a_cur[j] = *(const uint4*)(pa0 + j * 16);
  async_wait0();
  __syncthreads();

  const int nch = Ksize / KC;
  for (int ch = 0; ch < nch; ++ch) {
    const int cur = ch & 1;
    if (ch + 1 < nch) {
      async_copy_B(Bt, ldb, col0, (ch + 1) * KC, ldsB + (cur ^ 1) * LDSB_ELEMS);
#pragma unroll
      for (int j = 0; j < 4; ++j)
        a_nxt[j] = *(const uint4*)(pa0 + (ch + 1) * KC + j * 16);
    }
    const bf16_t* lb = ldsB + cur * LDSB_ELEMS;

    // Preload all 8 B fragments of this chunk (16 ds_load_b128 in flight),
    // then fire the 8 WMMAs with progressively draining dscnt waits.
    union uf { uint4 u[2]; v16bf v; };
    uf ub[8];
#pragma unroll
    for (int ks = 0; ks < 2; ++ks)
#pragma unroll
      for (int s = 0; s < 4; ++s) {
        const bf16_t* pb = lb + (s * 16 + l16) * BSTRIDE + ks * 32 + half * 8;
        ub[ks * 4 + s].u[0] = *(const uint4*)(pb);
        ub[ks * 4 + s].u[1] = *(const uint4*)(pb + 16);
      }
#pragma unroll
    for (int ks = 0; ks < 2; ++ks) {
      union uf ua;
      ua.u[0] = a_cur[ks * 2 + 0];
      ua.u[1] = a_cur[ks * 2 + 1];
#pragma unroll
      for (int s = 0; s < 4; ++s)
        acc[s] = __builtin_amdgcn_wmma_f32_16x16x32_bf16(
            false, ua.v, false, ub[ks * 4 + s].v, (short)0, acc[s], false, false);
    }

    if (ch + 1 < nch) {
#pragma unroll
      for (int j = 0; j < 4; ++j) a_cur[j] = a_nxt[j];
      async_wait0();      // next chunk's async LDS writes complete
      __syncthreads();    // visible to all waves; cur buffer free to overwrite
    }
  }
}
// C/D layout: lane holds col = col0 + s*16 + (lane&15); vgpr r holds
// row = row0 + r + 8*(lane>>4).

// ---------------------------------------------------------------------------
// Prep: convert weights to bf16, replicate mu^T (k,c) per batch.
// ---------------------------------------------------------------------------
__global__ __launch_bounds__(256) void k_prep(
    const float* __restrict__ W1, const float* __restrict__ W2,
    const float* __restrict__ mu,
    bf16_t* __restrict__ W1b, bf16_t* __restrict__ W2b,
    bf16_t* __restrict__ mT16) {
  int i = blockIdx.x * 256 + threadIdx.x;
  const int nW = CC * CC;
  if (i < nW) { W1b[i] = f2bf(W1[i]); W2b[i] = f2bf(W2[i]); }
  if (i < KK * CC) {
    int k = i / CC, c = i % CC;
    bf16_t v = f2bf(mu[(size_t)c * KK + k]);
#pragma unroll
    for (int b = 0; b < BB; ++b)
      mT16[((size_t)b * KK + k) * CC + c] = v;
  }
}

// ---------------------------------------------------------------------------
// f32 -> bf16 transpose (32x32 LDS tiles): (b,R,Cn) -> (b,Cn,R)
// ---------------------------------------------------------------------------
__global__ __launch_bounds__(256) void k_t32(const float* __restrict__ src,
                                             bf16_t* __restrict__ dst,
                                             int R, int Cn) {
  __shared__ __align__(16) bf16_t tile[32][40];
  const int b = blockIdx.z;
  const float* S = src + (size_t)b * R * Cn;
  bf16_t* D = dst + (size_t)b * Cn * R;
  const int c0 = blockIdx.x * 32, r0 = blockIdx.y * 32;
  const int tx = threadIdx.x, ty = threadIdx.y;
  for (int i = ty; i < 32; i += 8)
    tile[i][tx] = f2bf(S[(size_t)(r0 + i) * Cn + c0 + tx]);
  __syncthreads();
  for (int i = ty; i < 32; i += 8)
    D[(size_t)(c0 + i) * R + r0 + tx] = tile[tx][i];
}

// bf16 -> bf16 transpose; tile filled with async global->LDS b128 copies.
__global__ __launch_bounds__(256) void k_t16(const bf16_t* __restrict__ src,
                                             bf16_t* __restrict__ dst,
                                             int R, int Cn) {
  __shared__ __align__(16) bf16_t tile[32][40];
  const int b = blockIdx.z;
  const bf16_t* S = src + (size_t)b * R * Cn;
  bf16_t* D = dst + (size_t)b * Cn * R;
  const int c0 = blockIdx.x * 32, r0 = blockIdx.y * 32;
  const int tid = threadIdx.y * 32 + threadIdx.x;
  if (tid < 128) {  // 32 rows x 4 x 16B segments, one async op each
    const int row = tid >> 2, seg = tid & 3;
    const bf16_t* g = S + (size_t)(r0 + row) * Cn + c0 + seg * 8;
    unsigned l = lds_off32(&tile[row][seg * 8]);
    asm volatile("global_load_async_to_lds_b128 %0, %1, off"
                 :: "v"(l), "v"(g) : "memory");
  }
  async_wait0();
  __syncthreads();
  const int tx = threadIdx.x, ty = threadIdx.y;
  for (int i = ty; i < 32; i += 8)
    D[(size_t)(c0 + i) * R + r0 + tx] = tile[tx][i];
}

// ---------------------------------------------------------------------------
// GEMM 1: x' = W1 @ x + b1   -> Ax (b,c,n) bf16
// ---------------------------------------------------------------------------
__global__ __launch_bounds__(256) void k_conv1(
    const bf16_t* __restrict__ W1b, const bf16_t* __restrict__ Xt,
    const float* __restrict__ bias, bf16_t* __restrict__ Ax) {
  __shared__ __align__(16) bf16_t ldsB[2 * LDSB_ELEMS];
  const int b = blockIdx.z;
  const int wave = threadIdx.x >> 5, lane = threadIdx.x & 31;
  const int row0 = blockIdx.x * 128 + wave * 16;
  const int col0 = blockIdx.y * 64;
  v8f acc[4];
#pragma unroll
  for (int i = 0; i < 4; ++i) acc[i] = vzero8();
  gemm_block(W1b, CC, Xt + (size_t)b * NN * CC, CC, CC, row0, col0, acc, ldsB);
  const int half = lane >> 4, l16 = lane & 15;
  bf16_t* out = Ax + (size_t)b * CC * NN;
#pragma unroll
  for (int r = 0; r < 8; ++r) {
    const int row = row0 + r + 8 * half;
    const float bv = bias[row];
#pragma unroll
    for (int s = 0; s < 4; ++s)
      out[(size_t)row * NN + col0 + s * 16 + l16] = f2bf(acc[s][r] + bv);
  }
}

// ---------------------------------------------------------------------------
// GEMM 2: scores = x'^T @ m   (M=n, N=k, K=c) -> f32 scores (b,n,64)
// ---------------------------------------------------------------------------
__global__ __launch_bounds__(256) void k_scores(
    const bf16_t* __restrict__ Axt, const bf16_t* __restrict__ mT16,
    float* __restrict__ scores) {
  __shared__ __align__(16) bf16_t ldsB[2 * LDSB_ELEMS];
  const int b = blockIdx.z;
  const int wave = threadIdx.x >> 5, lane = threadIdx.x & 31;
  const int row0 = blockIdx.x * 128 + wave * 16;
  v8f acc[4];
#pragma unroll
  for (int i = 0; i < 4; ++i) acc[i] = vzero8();
  gemm_block(Axt + (size_t)b * NN * CC, CC, mT16 + (size_t)b * KK * CC, CC,
             CC, row0, 0, acc, ldsB);
  const int half = lane >> 4, l16 = lane & 15;
  float* out = scores + (size_t)b * NN * KK;
#pragma unroll
  for (int r = 0; r < 8; ++r)
#pragma unroll
    for (int s = 0; s < 4; ++s)
      out[(size_t)(row0 + r + 8 * half) * KK + s * 16 + l16] = acc[s][r];
}

// ---------------------------------------------------------------------------
// Softmax over k=64 per row; one wave per row. Writes z (n,k) and z^T (k,n).
// ---------------------------------------------------------------------------
__global__ __launch_bounds__(256) void k_softmax(
    const float* __restrict__ scores, bf16_t* __restrict__ z16,
    bf16_t* __restrict__ z16T) {
  const int wave = threadIdx.x >> 5, lane = threadIdx.x & 31;
  const size_t row = (size_t)blockIdx.x * 8 + wave;  // 0 .. BB*NN-1
  const float* s = scores + row * KK;
  float v0 = s[lane], v1 = s[lane + 32];
  float mx = fmaxf(v0, v1);
#pragma unroll
  for (int off = 16; off; off >>= 1) mx = fmaxf(mx, __shfl_xor(mx, off, 32));
  float e0 = __expf(v0 - mx), e1 = __expf(v1 - mx);
  float sum = e0 + e1;
#pragma unroll
  for (int off = 16; off; off >>= 1) sum += __shfl_xor(sum, off, 32);
  const float inv = 1.f / sum;
  const bf16_t b0 = f2bf(e0 * inv), b1 = f2bf(e1 * inv);
  z16[row * KK + lane] = b0;
  z16[row * KK + lane + 32] = b1;
  const int b = (int)(row >> 12), n = (int)(row & (NN - 1));
  bf16_t* zt = z16T + (size_t)b * KK * NN;
  zt[(size_t)lane * NN + n] = b0;
  zt[(size_t)(lane + 32) * NN + n] = b1;
}

// ---------------------------------------------------------------------------
// GEMM 3: m_hat = x' @ z   (M=c, N=k, K=n) -> f32 mhat (b,c,64)
// ---------------------------------------------------------------------------
__global__ __launch_bounds__(256) void k_mhat(
    const bf16_t* __restrict__ Ax, const bf16_t* __restrict__ z16T,
    float* __restrict__ mhat) {
  __shared__ __align__(16) bf16_t ldsB[2 * LDSB_ELEMS];
  const int b = blockIdx.z;
  const int wave = threadIdx.x >> 5, lane = threadIdx.x & 31;
  const int row0 = blockIdx.x * 128 + wave * 16;
  v8f acc[4];
#pragma unroll
  for (int i = 0; i < 4; ++i) acc[i] = vzero8();
  gemm_block(Ax + (size_t)b * CC * NN, NN, z16T + (size_t)b * KK * NN, NN,
             NN, row0, 0, acc, ldsB);
  const int half = lane >> 4, l16 = lane & 15;
  float* out = mhat + (size_t)b * CC * KK;
#pragma unroll
  for (int r = 0; r < 8; ++r)
#pragma unroll
    for (int s = 0; s < 4; ++s)
      out[(size_t)(row0 + r + 8 * half) * KK + s * 16 + l16] = acc[s][r];
}

// ---------------------------------------------------------------------------
// l2-normalize columns of m_hat over c; writes m (c,k) and m^T (k,c) bf16.
// ---------------------------------------------------------------------------
__global__ __launch_bounds__(256) void k_mnorm(
    const float* __restrict__ mhat, bf16_t* __restrict__ m16,
    bf16_t* __restrict__ mT16) {
  const int wave = threadIdx.x >> 5, lane = threadIdx.x & 31;
  const int col = blockIdx.x * 8 + wave;  // 0 .. BB*KK-1
  const int b = col >> 6, k = col & (KK - 1);
  const float* M = mhat + (size_t)b * CC * KK;
  float ss = 0.f;
  for (int c = lane; c < CC; c += 32) {
    float v = M[(size_t)c * KK + k];
    ss += v * v;
  }
#pragma unroll
  for (int off = 16; off; off >>= 1) ss += __shfl_xor(ss, off, 32);
  const float inv = 1.f / (1e-6f + sqrtf(ss));
  bf16_t* mo = m16 + (size_t)b * CC * KK;
  bf16_t* mt = mT16 + (size_t)b * KK * CC;
  for (int c = lane; c < CC; c += 32) {
    bf16_t v = f2bf(M[(size_t)c * KK + k] * inv);
    mo[(size_t)c * KK + k] = v;
    mt[(size_t)k * CC + c] = v;
  }
}

// ---------------------------------------------------------------------------
// GEMM 4: rec^T = relu(z @ m^T)   (M=n, N=c, K=k) -> recT (b,n,c) bf16
// ---------------------------------------------------------------------------
__global__ __launch_bounds__(256) void k_rec(
    const bf16_t* __restrict__ z16, const bf16_t* __restrict__ m16,
    bf16_t* __restrict__ recT) {
  __shared__ __align__(16) bf16_t ldsB[2 * LDSB_ELEMS];
  const int b = blockIdx.z;
  const int wave = threadIdx.x >> 5, lane = threadIdx.x & 31;
  const int row0 = blockIdx.x * 128 + wave * 16;
  const int col0 = blockIdx.y * 64;
  v8f acc[4];
#pragma unroll
  for (int i = 0; i < 4; ++i) acc[i] = vzero8();
  gemm_block(z16 + (size_t)b * NN * KK, KK, m16 + (size_t)b * CC * KK, KK,
             KK, row0, col0, acc, ldsB);
  const int half = lane >> 4, l16 = lane & 15;
  bf16_t* out = recT + (size_t)b * NN * CC;
#pragma unroll
  for (int r = 0; r < 8; ++r)
#pragma unroll
    for (int s = 0; s < 4; ++s)
      out[(size_t)(row0 + r + 8 * half) * CC + col0 + s * 16 + l16] =
          f2bf(fmaxf(acc[s][r], 0.f));
}

// ---------------------------------------------------------------------------
// GEMM 5: out = BN(W2 @ rec)   (M=o, N=n, K=c) -> f32 (b,c,h,w)
// ---------------------------------------------------------------------------
__global__ __launch_bounds__(256) void k_conv2(
    const bf16_t* __restrict__ W2b, const bf16_t* __restrict__ recT,
    const float* __restrict__ gamma, const float* __restrict__ beta,
    const float* __restrict__ mean, const float* __restrict__ var,
    float* __restrict__ out) {
  __shared__ __align__(16) bf16_t ldsB[2 * LDSB_ELEMS];
  const int b = blockIdx.z;
  const int wave = threadIdx.x >> 5, lane = threadIdx.x & 31;
  const int row0 = blockIdx.x * 128 + wave * 16;
  const int col0 = blockIdx.y * 64;
  v8f acc[4];
#pragma unroll
  for (int i = 0; i < 4; ++i) acc[i] = vzero8();
  gemm_block(W2b, CC, recT + (size_t)b * NN * CC, CC, CC, row0, col0, acc, ldsB);
  const int half = lane >> 4, l16 = lane & 15;
  float* O = out + (size_t)b * CC * NN;
#pragma unroll
  for (int r = 0; r < 8; ++r) {
    const int o = row0 + r + 8 * half;
    const float inv = gamma[o] * rsqrtf(var[o] + 1e-5f);
    const float sh = beta[o] - mean[o] * inv;
#pragma unroll
    for (int s = 0; s < 4; ++s)
      O[(size_t)o * NN + col0 + s * 16 + l16] = acc[s][r] * inv + sh;
  }
}

// ---------------------------------------------------------------------------
extern "C" void kernel_launch(void* const* d_in, const int* in_sizes, int n_in,
                              void* d_out, int out_size, void* d_ws,
                              size_t ws_size, hipStream_t stream) {
  (void)in_sizes; (void)n_in; (void)out_size; (void)ws_size;
  const float* x     = (const float*)d_in[0];
  const float* W1    = (const float*)d_in[1];
  const float* b1    = (const float*)d_in[2];
  const float* mu    = (const float*)d_in[3];
  const float* W2    = (const float*)d_in[4];
  const float* gamma = (const float*)d_in[5];
  const float* beta  = (const float*)d_in[6];
  const float* mean  = (const float*)d_in[7];
  const float* var   = (const float*)d_in[8];
  float* out = (float*)d_out;

  char* p = (char*)d_ws;
  bf16_t* W1b  = (bf16_t*)p; p += (size_t)CC * CC * 2;
  bf16_t* W2b  = (bf16_t*)p; p += (size_t)CC * CC * 2;
  bf16_t* XT   = (bf16_t*)p; p += (size_t)BB * NN * CC * 2;  // x^T, reused as x'^T
  bf16_t* Ax   = (bf16_t*)p; p += (size_t)BB * CC * NN * 2;  // x' (c,n)
  bf16_t* mT16 = (bf16_t*)p; p += (size_t)BB * KK * CC * 2;  // m^T (k,c)
  bf16_t* m16  = (bf16_t*)p; p += (size_t)BB * CC * KK * 2;  // m (c,k)
  float*  mhat = (float*)p;  p += (size_t)BB * CC * KK * 4;
  float*  sc   = (float*)p;  p += (size_t)BB * NN * KK * 4;
  bf16_t* z16  = (bf16_t*)p; p += (size_t)BB * NN * KK * 2;  // z (n,k)
  bf16_t* z16T = (bf16_t*)p; p += (size_t)BB * KK * NN * 2;  // z^T (k,n)
  bf16_t* recT = (bf16_t*)p; p += (size_t)BB * NN * CC * 2;  // rec^T (n,c)
  bf16_t* Axt  = XT;  // x^T dead after conv1; reuse buffer for x'^T

  k_prep<<<dim3(1024), 256, 0, stream>>>(W1, W2, mu, W1b, W2b, mT16);
  k_t32<<<dim3(NN / 32, CC / 32, BB), dim3(32, 8), 0, stream>>>(x, XT, CC, NN);
  k_conv1<<<dim3(CC / 128, NN / 64, BB), 256, 0, stream>>>(W1b, XT, b1, Ax);
  k_t16<<<dim3(NN / 32, CC / 32, BB), dim3(32, 8), 0, stream>>>(Ax, Axt, CC, NN);

  for (int s = 0; s < STAGES; ++s) {
    k_scores<<<dim3(NN / 128, 1, BB), 256, 0, stream>>>(Axt, mT16, sc);
    k_softmax<<<dim3(BB * NN / 8), 256, 0, stream>>>(sc, z16, z16T);
    k_mhat<<<dim3(CC / 128, 1, BB), 256, 0, stream>>>(Ax, z16T, mhat);
    k_mnorm<<<dim3(BB * KK / 8), 256, 0, stream>>>(mhat, m16, mT16);
  }

  k_rec<<<dim3(NN / 128, CC / 64, BB), 256, 0, stream>>>(z16, m16, recT);
  k_conv2<<<dim3(CC / 128, NN / 64, BB), 256, 0, stream>>>(W2b, recT, gamma,
                                                           beta, mean, var, out);
}
